// MultiHeadAttentionWithKVCache_46291157516527
// MI455X (gfx1250) — matrix operand order repaired and checked by reading, not compiled, not run on hardware
//
#include <hip/hip_runtime.h>
#include <hip/hip_bf16.h>

#define B_    16
#define S_    16
#define D_    2048
#define H_    16
#define HD_   128
#define PAST_ 4096
#define TOT_  4112
#define SCALE_ 0.088388347648318447f   // 1/sqrt(128)
#define NW_   8                        // waves per attention block (key split)

typedef __attribute__((ext_vector_type(16))) __bf16 v16bf;
typedef __attribute__((ext_vector_type(8)))  float  v8f;
typedef __attribute__((ext_vector_type(4)))  float  v4f;   // native vector: OK for NT builtins

static __device__ __forceinline__ v4f ldg4(const float* p) {
  return *reinterpret_cast<const v4f*>(p);
}
// streaming (non-temporal) variants for the read-once / write-once KV streams
static __device__ __forceinline__ v4f ldg4_nt(const float* p) {
  return __builtin_nontemporal_load(reinterpret_cast<const v4f*>(p));
}
static __device__ __forceinline__ void stg4_nt(float* p, v4f v) {
  __builtin_nontemporal_store(v, reinterpret_cast<v4f*>(p));
}

// native f32 -> bf16 conversion (compiler should emit v_cvt_*bf16_f32)
static __device__ __forceinline__ v16bf pack16(v4f a, v4f b, v4f c, v4f d) {
  v16bf r;
  r[0]  = (__bf16)a[0]; r[1]  = (__bf16)a[1]; r[2]  = (__bf16)a[2]; r[3]  = (__bf16)a[3];
  r[4]  = (__bf16)b[0]; r[5]  = (__bf16)b[1]; r[6]  = (__bf16)b[2]; r[7]  = (__bf16)b[3];
  r[8]  = (__bf16)c[0]; r[9]  = (__bf16)c[1]; r[10] = (__bf16)c[2]; r[11] = (__bf16)c[3];
  r[12] = (__bf16)d[0]; r[13] = (__bf16)d[1]; r[14] = (__bf16)d[2]; r[15] = (__bf16)d[3];
  return r;
}

static __device__ __forceinline__ v8f wmma_bf16(v16bf a, v16bf b, v8f c) {
  return __builtin_amdgcn_wmma_f32_16x16x32_bf16(false, a, false, b, (short)0, c, false, false);
}

// ---------------------------------------------------------------------------
// GEMM: out = A[256x2048] @ W^T + bias.  One wave -> 16x16 tile, K-loop of 64
// wmma_f32_16x16x32_bf16.  MODE 0: scatter to Q_ws [B,H,S,HD]; MODE 1: scatter
// to k/v out rows PAST+s; MODE 2: row-major [256,2048].
// ---------------------------------------------------------------------------
template <int MODE>
__global__ __launch_bounds__(32) void gemm_bf16_wmma(
    const float* __restrict__ A, const float* __restrict__ W,
    const float* __restrict__ bias, float* __restrict__ out) {
  const int lane = threadIdx.x;
  const int ln   = lane & 15;
  const int half = lane >> 4;
  const int n0 = blockIdx.x * 16;
  const int m0 = blockIdx.y * 16;

  const float* ap = A + (size_t)(m0 + ln) * D_ + half * 16;
  const float* wp = W + (size_t)(n0 + ln) * D_ + half * 16;

  v8f acc = {0.f, 0.f, 0.f, 0.f, 0.f, 0.f, 0.f, 0.f};
  for (int kk = 0; kk < D_; kk += 32) {
    v16bf av = pack16(ldg4(ap + kk),      ldg4(ap + kk + 4),
                      ldg4(ap + kk + 8), ldg4(ap + kk + 12));
    v16bf bv = pack16(ldg4(wp + kk),     ldg4(wp + kk + 4),
                      ldg4(wp + kk + 8), ldg4(wp + kk + 12));
    acc = wmma_bf16(av, bv, acc);
  }

  const float bn = bias[n0 + ln];
#pragma unroll
  for (int i = 0; i < 8; ++i) {
    const int m = m0 + i + half * 8;
    const int n = n0 + ln;
    const float val = acc[i] + bn;
    size_t idx;
    if (MODE == 0) {
      const int bb = m >> 4, s = m & 15, hh = n >> 7, hd = n & 127;
      idx = ((size_t)(bb * H_ + hh) * S_ + s) * HD_ + hd;
    } else if (MODE == 1) {
      const int bb = m >> 4, s = m & 15, hh = n >> 7, hd = n & 127;
      idx = ((size_t)(bb * H_ + hh) * TOT_ + PAST_ + s) * HD_ + hd;
    } else {
      idx = (size_t)m * D_ + n;
    }
    out[idx] = val;
  }
}

// ---------------------------------------------------------------------------
// Flash attention, 8-way key-split per (b,h), fused cache->output copy.
// Main tiles (t=0..127): source is always the cache, copy unconditional,
// NT loads/stores (streamed once, keep L2 for GEMM weights).  Tail tile
// (new tokens) is a second inlined instantiation with clamping.
// ---------------------------------------------------------------------------
__global__ __launch_bounds__(256) void flash_attn_copy(
    const float* __restrict__ q_ws,
    const float* __restrict__ cache_k,
    const float* __restrict__ cache_v,
    const int*   __restrict__ amask,
    float* __restrict__ k_out,
    float* __restrict__ v_out,
    float* __restrict__ ctx) {
  __shared__ float plds[NW_][16][33];    // per-wave P tile (16x32, padded)
  __shared__ float o_lds[NW_][16][129];  // per-wave O partial (16x128, padded)
  __shared__ float m_lds[NW_][16];
  __shared__ float l_lds[NW_][16];

  const int tid  = threadIdx.x;
  const int wid  = tid >> 5;
  const int lane = tid & 31;
  const int ln   = lane & 15;
  const int half = lane >> 4;
  const int bh = blockIdx.x;             // b*H + h
  const int b  = bh >> 4;
  const int h  = bh & 15;

  const float* kcache = cache_k + (size_t)bh * PAST_ * HD_;
  const float* vcache = cache_v + (size_t)bh * PAST_ * HD_;
  float* kdst_base = k_out + (size_t)bh * TOT_ * HD_;
  float* vdst_base = v_out + (size_t)bh * TOT_ * HD_;

  // Q A-fragments (K-dim 128 = 4 chunks of 32)
  v16bf aq[4];
  {
    const float* qp = q_ws + ((size_t)bh * S_ + ln) * HD_;
#pragma unroll
    for (int c = 0; c < 4; ++c) {
      const int kb = c * 32 + half * 16;
      aq[c] = pack16(ldg4(qp + kb), ldg4(qp + kb + 4),
                     ldg4(qp + kb + 8), ldg4(qp + kb + 12));
    }
  }

  v8f o[8];
#pragma unroll
  for (int j = 0; j < 8; ++j) o[j] = (v8f){0.f,0.f,0.f,0.f,0.f,0.f,0.f,0.f};
  float mrun[8], lsum[8];
  int moff[8];
#pragma unroll
  for (int i = 0; i < 8; ++i) {
    mrun[i] = -1e20f; lsum[i] = 0.f;
    moff[i] = (b * S_ + i + half * 8) * TOT_;   // mask row offsets
  }

  // One 32-key step.  docopy/clamp are literal at both call sites -> two
  // straight-line inlined bodies, no per-element pointer selection.
  auto step = [&](int kpos0, const float* ksrc, const float* vsrc,
                  float* kdst, float* vdst, bool docopy, bool clamp) {
    float a0s[8], a1s[8];
    // ---- scores: two 16-key sub-tiles ----
#pragma unroll
    for (int sub = 0; sub < 2; ++sub) {
      const int rl  = sub * 16 + ln;                       // local key row
      const int rle = clamp ? (rl < 16 ? rl : 15) : rl;    // tail clamp
      const float* src = ksrc + (size_t)rle * HD_;
      float* dst = kdst + (size_t)rl * HD_;
      v8f s = {0.f,0.f,0.f,0.f,0.f,0.f,0.f,0.f};
#pragma unroll
      for (int c = 0; c < 4; ++c) {
        const int hd0 = c * 32 + half * 16;
        v4f k0, k1, k2, k3;
        if (docopy) {           // streaming path: NT load + NT copy store
          k0 = ldg4_nt(src + hd0);     k1 = ldg4_nt(src + hd0 + 4);
          k2 = ldg4_nt(src + hd0 + 8); k3 = ldg4_nt(src + hd0 + 12);
          stg4_nt(dst + hd0, k0);      stg4_nt(dst + hd0 + 4, k1);
          stg4_nt(dst + hd0 + 8, k2);  stg4_nt(dst + hd0 + 12, k3);
        } else {
          k0 = ldg4(src + hd0);        k1 = ldg4(src + hd0 + 4);
          k2 = ldg4(src + hd0 + 8);    k3 = ldg4(src + hd0 + 12);
        }
        s = wmma_bf16(aq[c], pack16(k0, k1, k2, k3), s);
      }
      const int kg  = kpos0 + rl;
      const int kgc = clamp ? (kg < TOT_ ? kg : TOT_ - 1) : kg;
#pragma unroll
      for (int i = 0; i < 8; ++i) {
        const int mv = amask[moff[i] + kgc];
        const bool dead = (clamp && kg >= TOT_) || (mv == 0);
        const float sc = dead ? -1e20f : s[i] * SCALE_;
        if (sub == 0) a0s[i] = sc; else a1s[i] = sc;
      }
    }
    // ---- online softmax (rows across 16-lane halves) ----
#pragma unroll
    for (int i = 0; i < 8; ++i) {
      float rmax = fmaxf(a0s[i], a1s[i]);
#pragma unroll
      for (int x = 1; x < 16; x <<= 1) rmax = fmaxf(rmax, __shfl_xor(rmax, x, 32));
      const float mnew = fmaxf(mrun[i], rmax);
      const float e0 = __expf(a0s[i] - mnew);
      const float e1 = __expf(a1s[i] - mnew);
      float rs = e0 + e1;
#pragma unroll
      for (int x = 1; x < 16; x <<= 1) rs += __shfl_xor(rs, x, 32);
      const float alpha = __expf(mrun[i] - mnew);
      lsum[i] = lsum[i] * alpha + rs;
      mrun[i] = mnew;
      a0s[i] = e0; a1s[i] = e1;                 // reuse as p-values
#pragma unroll
      for (int j = 0; j < 8; ++j) o[j][i] *= alpha;
    }
    // ---- P (C-layout) -> per-wave LDS -> A-fragment ----
#pragma unroll
    for (int i = 0; i < 8; ++i) {
      const int m = i + half * 8;
      plds[wid][m][ln]      = a0s[i];
      plds[wid][m][16 + ln] = a1s[i];
    }
    asm volatile("s_wait_dscnt 0x0" ::: "memory");  // wave-internal DS ordering
    v16bf pa;
    {
      const int kb = half * 16;
#pragma unroll
      for (int e = 0; e < 8; ++e) pa[e]     = (__bf16)plds[wid][ln][kb + e];
#pragma unroll
      for (int e = 0; e < 8; ++e) pa[8 + e] = (__bf16)plds[wid][ln][kb + 8 + e];
    }
    asm volatile("s_wait_dscnt 0x0" ::: "memory");
    // ---- O += P @ V (8 N-chunks), fused V copy ----
#pragma unroll
    for (int j = 0; j < 8; ++j) {
      v16bf bu;
#pragma unroll
      for (int r = 0; r < 16; ++r) {
        const int rl  = half * 16 + r;
        const int rle = clamp ? (rl < 16 ? rl : 15) : rl;
        float val;
        if (docopy) {
          val = __builtin_nontemporal_load(vsrc + (size_t)rle * HD_ + j * 16 + ln);
          __builtin_nontemporal_store(val, vdst + (size_t)rl * HD_ + j * 16 + ln);
        } else {
          val = vsrc[(size_t)rle * HD_ + j * 16 + ln];
        }
        bu[r] = (__bf16)val;
      }
      o[j] = wmma_bf16(pa, bu, o[j]);
    }
  };

  // key-split loop: wave wid takes tiles t = wid, wid+8, ...
  for (int t = wid; t < 129; t += NW_) {
    const size_t off = (size_t)t * 32 * HD_;
    if (t < 128) {
      step(t * 32, kcache + off, vcache + off,
           kdst_base + off, vdst_base + off, true, false);
    } else {
      step(t * 32, kdst_base + off, vdst_base + off,
           nullptr, nullptr, false, true);
    }
  }

  // ---- dump partials and combine across the 8 waves ----
#pragma unroll
  for (int i = 0; i < 8; ++i) {
    const int m = i + half * 8;
    if (ln == 0) { m_lds[wid][m] = mrun[i]; l_lds[wid][m] = lsum[i]; }
  }
#pragma unroll
  for (int j = 0; j < 8; ++j)
#pragma unroll
    for (int i = 0; i < 8; ++i)
      o_lds[wid][i + half * 8][j * 16 + ln] = o[j][i];
  __syncthreads();

  for (int e = tid; e < 16 * 128; e += 256) {
    const int m = e >> 7, col = e & 127;
    float gm = -3.0e38f;
#pragma unroll
    for (int w = 0; w < NW_; ++w) gm = fmaxf(gm, m_lds[w][m]);
    float den = 0.f, num = 0.f;
#pragma unroll
    for (int w = 0; w < NW_; ++w) {
      const float sc = __expf(m_lds[w][m] - gm);
      den += l_lds[w][m] * sc;
      num += o_lds[w][m][col] * sc;
    }
    ctx[((size_t)(b * S_ + m)) * D_ + h * HD_ + col] = num / den;
  }
}

// ---------------------------------------------------------------------------
extern "C" void kernel_launch(void* const* d_in, const int* in_sizes, int n_in,
                              void* d_out, int out_size, void* d_ws, size_t ws_size,
                              hipStream_t stream) {
  const float* X  = (const float*)d_in[0];
  const int*   am = (const int*)  d_in[1];
  const float* ck = (const float*)d_in[2];
  const float* cv = (const float*)d_in[3];
  const float* Wq = (const float*)d_in[4];
  const float* bq = (const float*)d_in[5];
  const float* Wk = (const float*)d_in[6];
  const float* bk = (const float*)d_in[7];
  const float* Wv = (const float*)d_in[8];
  const float* bv = (const float*)d_in[9];
  const float* Wo = (const float*)d_in[10];
  const float* bo = (const float*)d_in[11];

  float* out  = (float*)d_out;
  float* kout = out  + (size_t)B_ * S_ * D_;
  float* vout = kout + (size_t)B_ * H_ * TOT_ * HD_;

  float* q_ws = (float*)d_ws;
  float* ctx  = q_ws + (size_t)B_ * H_ * S_ * HD_;

  const dim3 gblk(32, 1, 1);
  const dim3 ggrd(D_ / 16, (B_ * S_) / 16, 1);

  gemm_bf16_wmma<0><<<ggrd, gblk, 0, stream>>>(X, Wq, bq, q_ws);
  gemm_bf16_wmma<1><<<ggrd, gblk, 0, stream>>>(X, Wk, bk, kout);
  gemm_bf16_wmma<1><<<ggrd, gblk, 0, stream>>>(X, Wv, bv, vout);
  flash_attn_copy<<<dim3(B_ * H_), dim3(256, 1, 1), 0, stream>>>(
      q_ws, ck, cv, am, kout, vout, ctx);
  gemm_bf16_wmma<2><<<ggrd, gblk, 0, stream>>>(ctx, Wo, bo, out);
}
